// VectorQuantizer_82463372083397
// MI455X (gfx1250) — compile-verified, matrix operand order checked by reading, not run on hardware
//
#include <hip/hip_runtime.h>
#include <hip/hip_bf16.h>
#include <math.h>

typedef _Float16 v16h __attribute__((ext_vector_type(16)));
typedef _Float16 v8h  __attribute__((ext_vector_type(8)));
typedef float    v8f  __attribute__((ext_vector_type(8)));

#define NUM_EMB 1024
#define EMB_DIM 256
#define BATCH   64
#define HW      1024                 // 32*32 pixels per image
#define NTOT    (BATCH * HW)         // 65536 vectors
#define TILE_P  128                  // pixels per block
#define CHUNK_K 128                  // codes per LDS chunk
#define LDA     (EMB_DIM + 8)        // padded f16 row stride (528 B, 16B-aligned)
#define QSIZE   (BATCH * EMB_DIM * HW)  // 16777216

// workspace layout (bytes)
#define WS_WN_F32   0
#define WS_WN_F16   (NUM_EMB * EMB_DIM * 4)
#define WS_IDX      (WS_WN_F16 + NUM_EMB * EMB_DIM * 2)
#define WS_PART     (WS_IDX + NTOT * 4)

// ---------------------------------------------------------------------------
// Kernel 1: normalize codebook rows -> f32 and f16 copies in workspace
// ---------------------------------------------------------------------------
__global__ void vq_prep(const float* __restrict__ w,
                        float* __restrict__ wn32,
                        _Float16* __restrict__ wn16) {
    const int k = blockIdx.x;          // codebook row
    const int t = threadIdx.x;         // dim
    float v  = w[k * EMB_DIM + t];
    float ss = v * v;
    #pragma unroll
    for (int m = 16; m >= 1; m >>= 1) ss += __shfl_xor(ss, m, 32);
    __shared__ float red[8];
    if ((t & 31) == 0) red[t >> 5] = ss;
    __syncthreads();
    float tot = 0.f;
    #pragma unroll
    for (int i = 0; i < 8; ++i) tot += red[i];
    const float rn = 1.f / fmaxf(sqrtf(tot), 1e-12f);
    const float nv = v * rn;
    wn32[k * EMB_DIM + t] = nv;
    wn16[k * EMB_DIM + t] = (_Float16)nv;
}

// ---------------------------------------------------------------------------
// Kernel 2: normalize input tile, WMMA score sweep over all 1024 codes.
//   - codebook chunks staged with GLOBAL_LOAD_ASYNC_TO_LDS_B128 (ASYNCcnt)
//   - B fragments batched per 16x16 tile so the scheduler can overlap the
//     next tile's ds_load_b128 with the current 8-deep WMMA chain
// ---------------------------------------------------------------------------
__global__ void __launch_bounds__(256) vq_main(const float* __restrict__ x,
                                               const _Float16* __restrict__ wn16,
                                               int*   __restrict__ idx_ws,
                                               float* __restrict__ out_idx_f,
                                               float* __restrict__ partials) {
    __shared__ __align__(16) _Float16 Ax[TILE_P * LDA];   // normalized pixels (f16)
    __shared__ __align__(16) _Float16 Bx[CHUNK_K * LDA];  // codebook chunk  (f16)
    __shared__ float ssq[256];
    __shared__ float rnorm[TILE_P];
    __shared__ float rowerr[TILE_P];

    const int tid = threadIdx.x;
    const int b   = blockIdx.x >> 3;            // image
    const int p0  = (blockIdx.x & 7) * TILE_P;  // first pixel of tile
    const float* xb = x + (size_t)b * EMB_DIM * HW + p0;

    // per-thread staging span for async B copies: 256 contiguous bytes
    const int srow = tid >> 1, shalf = tid & 1;
    const unsigned int lds_base =
        (unsigned int)(size_t)(&Bx[srow * LDA + shalf * 128]);   // low 32b = LDS offset

    // ---- stage + L2-normalize the A tile (coalesced: fixed d, contiguous p)
    {
        const int p  = tid & 127;  // pixel in tile
        const int dh = tid >> 7;   // which of 2 d-interleaves this thread owns
        float ss = 0.f;
        #pragma unroll 4
        for (int dd = 0; dd < EMB_DIM / 2; ++dd) {
            const int d = dd * 2 + dh;
            const float v = xb[(size_t)d * HW + p];
            ss += v * v;
            Ax[p * LDA + d] = (_Float16)v;
        }
        ssq[tid] = ss;
        __syncthreads();
        if (tid < TILE_P)
            rnorm[tid] = 1.f / fmaxf(sqrtf(ssq[tid] + ssq[tid + 128]), 1e-12f);
        __syncthreads();
        const float rn = rnorm[p];
        #pragma unroll 4
        for (int dd = 0; dd < EMB_DIM / 2; ++dd) {
            const int d = dd * 2 + dh;
            Ax[p * LDA + d] = (_Float16)((float)Ax[p * LDA + d] * rn);
        }
    }
    __syncthreads();

    // ---- per-wave A fragments (ISA 16-bit A layout: lane%16 = M,
    //      elems 0..7 -> K = 8*(lane/16)+0..7, elems 8..15 -> +16)
    const int lane = tid & 31;
    const int wv   = tid >> 5;      // wave -> 16-row block
    const int ln16 = lane & 15;
    const int g    = lane >> 4;
    const _Float16* arow = &Ax[(wv * 16 + ln16) * LDA];
    v16h afrag[8];
    #pragma unroll
    for (int c = 0; c < 8; ++c) {
        union { v16h v; v8h h[2]; } u;
        u.h[0] = *(const v8h*)(arow + c * 32 + g * 8);
        u.h[1] = *(const v8h*)(arow + c * 32 + 16 + g * 8);
        afrag[c] = u.v;
    }

    float bestv[8];
    int   besti[8];
    #pragma unroll
    for (int r = 0; r < 8; ++r) { bestv[r] = -3.0e38f; besti[r] = 0; }

    for (int ch = 0; ch < NUM_EMB / CHUNK_K; ++ch) {
        // ---- async-stage codebook chunk straight into LDS (no VGPR bounce)
        {
            const unsigned long long gsrc = (unsigned long long)(size_t)
                (wn16 + (size_t)(ch * CHUNK_K + srow) * EMB_DIM + shalf * 128);
            #pragma unroll
            for (int q = 0; q < 16; ++q) {
                asm volatile("global_load_async_to_lds_b128 %0, %1, off"
                             :
                             : "v"(lds_base + q * 16),
                               "v"(gsrc + (unsigned long long)(q * 16))
                             : "memory");
            }
            asm volatile("s_wait_asynccnt 0x0" ::: "memory");
        }
        __syncthreads();

        #pragma unroll
        for (int t = 0; t < 8; ++t) {           // 8 column tiles of 16 codes
            const _Float16* brow = &Bx[(t * 16 + ln16) * LDA];
            // batch all 16 ds_load_b128 for this tile, then run the WMMA chain
            v16h bf[8];
            #pragma unroll
            for (int c = 0; c < 8; ++c) {       // D = 256 -> 8 K-slices of 32
                union { v16h v; v8h h[2]; } u;
                u.h[0] = *(const v8h*)(brow + c * 32 + g * 8);
                u.h[1] = *(const v8h*)(brow + c * 32 + 16 + g * 8);
                bf[c] = u.v;
            }
            v8f acc = {};
            #pragma unroll
            for (int c = 0; c < 8; ++c)
                acc = __builtin_amdgcn_wmma_f32_16x16x32_f16(
                    false, afrag[c], false, bf[c], (short)0, acc, false, false);

            // C layout: lane holds N = ln16, rows M = r + 8*g
            const int code = ch * CHUNK_K + t * 16 + ln16;
            #pragma unroll
            for (int r = 0; r < 8; ++r) {
                const float v = acc[r];
                if (v > bestv[r]) { bestv[r] = v; besti[r] = code; }
            }
        }
        __syncthreads();
    }

    // ---- argmax butterfly over the 16-lane half (same M set, different N),
    //      min-index tie-break to match argmin first-occurrence
    #pragma unroll
    for (int m = 1; m <= 8; m <<= 1) {
        #pragma unroll
        for (int r = 0; r < 8; ++r) {
            const float ov = __shfl_xor(bestv[r], m, 32);
            const int   oi = __shfl_xor(besti[r], m, 32);
            if (ov > bestv[r] || (ov == bestv[r] && oi < besti[r])) {
                bestv[r] = ov; besti[r] = oi;
            }
        }
    }

    if (ln16 == 0) {   // lanes 0 (rows 0..7) and 16 (rows 8..15) of each wave
        #pragma unroll
        for (int r = 0; r < 8; ++r) {
            const int mrow = wv * 16 + g * 8 + r;   // row within tile
            const int n = b * HW + p0 + mrow;
            idx_ws[n]    = besti[r];
            out_idx_f[n] = (float)besti[r];
            rowerr[mrow] = 2.f - 2.f * bestv[r];    // ||q - fn||^2 (unit vectors)
        }
    }
    __syncthreads();
    if (tid == 0) {
        float s = 0.f;
        for (int i = 0; i < TILE_P; ++i) s += rowerr[i];  // fixed order: deterministic
        partials[blockIdx.x] = s;
    }
}

// ---------------------------------------------------------------------------
// Kernel 3: quant_st output = wn_f32[idx] scattered to [B,D,H,W]
// ---------------------------------------------------------------------------
__global__ void vq_scatter(const float* __restrict__ wn32,
                           const int* __restrict__ idx,
                           float* __restrict__ outq) {
    const size_t f = (size_t)blockIdx.x * blockDim.x + threadIdx.x;
    const int p = (int)(f & (HW - 1));
    const int d = (int)((f >> 10) & (EMB_DIM - 1));
    const int b = (int)(f >> 18);
    const int k = idx[b * HW + p];
    outq[f] = wn32[k * EMB_DIM + d];
}

// ---------------------------------------------------------------------------
// Kernel 4: loss = 1.25 * mean((quant - fn)^2), fixed-order final sum
// ---------------------------------------------------------------------------
__global__ void vq_loss(const float* __restrict__ partials, float* __restrict__ loss) {
    if (blockIdx.x == 0 && threadIdx.x == 0) {
        float s = 0.f;
        for (int i = 0; i < NTOT / TILE_P; ++i) s += partials[i];
        loss[0] = 1.25f * s / (float)QSIZE;
    }
}

// ---------------------------------------------------------------------------
extern "C" void kernel_launch(void* const* d_in, const int* in_sizes, int n_in,
                              void* d_out, int out_size, void* d_ws, size_t ws_size,
                              hipStream_t stream) {
    const float* x = (const float*)d_in[0];   // [64,256,32,32] f32
    const float* w = (const float*)d_in[1];   // [1024,256]     f32
    float* out = (float*)d_out;

    char* ws = (char*)d_ws;
    float*    wn32 = (float*)(ws + WS_WN_F32);
    _Float16* wn16 = (_Float16*)(ws + WS_WN_F16);
    int*      idxw = (int*)(ws + WS_IDX);
    float*    part = (float*)(ws + WS_PART);

    float* loss  = out;               // [1]
    float* quant = out + 1;           // [QSIZE]
    float* idxf  = out + 1 + QSIZE;   // [NTOT]

    vq_prep   <<<NUM_EMB, EMB_DIM, 0, stream>>>(w, wn32, wn16);
    vq_main   <<<NTOT / TILE_P, 256, 0, stream>>>(x, wn16, idxw, idxf, part);
    vq_scatter<<<QSIZE / 256, 256, 0, stream>>>(wn32, idxw, quant);
    vq_loss   <<<1, 32, 0, stream>>>(part, loss);
}
